// DifferentiableDTW_7928509629105
// MI455X (gfx1250) — matrix operand order repaired and checked by reading, not compiled, not run on hardware
//
#include <hip/hip_runtime.h>
#include <hip/hip_bf16.h>
#include <math.h>

// CDNA5 (gfx1250) fused softmax-GEMM for the differentiable-DTW reference.
// warped[b,i,f] = sum_j softmax_j(-|score[b,i]-template[j]|) * feature[b,j,f]
// l2 = 1e-7 * sqrt(sum(warped^2)).
//
// Fast path: pre-pass converts feature to f16 pre-swizzled into a contiguous-k (k = 16h + s)
// WMMA B-matrix layout (per batch/K-chunk tiles); main kernel double-buffers
// 32KB chunks into LDS with global_load_async_to_lds_b128 (ASYNCcnt-fenced),
// generates two exp() A-tiles (32 output rows) in LDS, and runs
// v_wmma_f32_16x16x32_f16 with each B tile reused for both M-tiles.

typedef __attribute__((ext_vector_type(16))) _Float16 v16h;
typedef __attribute__((ext_vector_type(8)))  float    v8f;

#define S_DIM   2048
#define F_DIM   512
#define B_DIM   8
#define KC      32                 // K (j) chunk per WMMA iteration
#define NCHUNK  (S_DIM / KC)       // 64
#define THREADS 256                // 8 waves (wave32)
#define WAVES   (THREADS / 32)
#define NT_PER_WAVE ((F_DIM / 16) / WAVES)   // 4 N-tiles per wave
#define CHUNK_HALVES (KC * F_DIM)            // 16384 f16 per chunk

#define MROWS   32                 // fast path: output rows per block
#define MTILES  (MROWS / 16)       // 2 A-tiles per chunk
#define ROWS_FB 16                 // fallback: output rows per block

#if __has_builtin(__builtin_amdgcn_s_wait_asynccnt)
#define WAIT_ASYNC0() __builtin_amdgcn_s_wait_asynccnt(0)
#else
#define WAIT_ASYNC0() asm volatile("s_wait_asynccnt 0" ::: "memory")
#endif

// ---------------------------------------------------------------------------
// Pre-pass: feature f32 -> f16 in WMMA-B tile layout.
// flat index = ((((b*64 + jc)*32 + t)*32 + l)*16 + h)
//   where for element (k,n) of chunk jc: t=n>>4, l=(n&15)+(k&16), h=k&15.
// One thread per (b,jc,t,l): reads 16 f32 (coalesced across threads),
// writes 32 contiguous bytes.
// ---------------------------------------------------------------------------
__global__ __launch_bounds__(256)
void dtw_convert_kernel(const float* __restrict__ feature,
                        _Float16* __restrict__ featH)
{
    const int tid  = blockIdx.x * 256 + threadIdx.x;   // 524288 total
    const int l    = tid & 31;
    const int t    = (tid >> 5) & 31;
    const int jc   = (tid >> 10) & 63;
    const int b    = tid >> 16;
    const int n    = (t << 4) + (l & 15);
    const int koff = l & 16;

    const float* src = feature
        + ((size_t)b * S_DIM + (size_t)(jc * KC + koff)) * F_DIM + n;
    v16h v;
    #pragma unroll
    for (int k = 0; k < 16; ++k)
        v[k] = (_Float16)src[(size_t)k * F_DIM];
    *(v16h*)(featH + ((size_t)tid << 4)) = v;
}

// ---------------------------------------------------------------------------
// Fast main kernel: async-LDS double-buffered WMMA, 32 rows per block.
// ---------------------------------------------------------------------------
__global__ __launch_bounds__(THREADS)
void dtw_warp_async_kernel(const float* __restrict__ score,
                           const _Float16* __restrict__ featH,
                           const float* __restrict__ tmpl,
                           float* __restrict__ out,
                           float* __restrict__ ss_accum)
{
    __shared__ float Tsh[S_DIM];                      // template (8 KB)
    __shared__ float Ssh[MROWS];
    __shared__ float Zsh[MROWS];
    __shared__ float Psh[THREADS];
    __shared__ __align__(32) _Float16 Ash[2][MTILES * 16 * KC];  // 2 x 2 KB
    __shared__ __align__(32) _Float16 Bsh[2][CHUNK_HALVES];      // 2 x 32 KB

    const int tid  = threadIdx.x;
    const int lane = tid & 31;
    const int w    = tid >> 5;
    const int b    = blockIdx.x >> 6;                 // 64 blocks per batch
    const int i0   = (blockIdx.x & 63) << 5;          // 32 rows per block

    const _Float16* gB = featH + (size_t)b * (NCHUNK * CHUNK_HALVES);

    // issue one chunk (32 KB) as 8 per-thread async b128 copies into LDS
    auto issue_async = [&](int buf, int c) {
        const unsigned lds0 =
            (unsigned)(size_t)(&Bsh[buf][0]) + (unsigned)(tid * 16);
        unsigned long long g0 =
            (unsigned long long)(size_t)(gB + (size_t)c * CHUNK_HALVES)
            + (unsigned long long)(tid * 16);
        #pragma unroll
        for (int i = 0; i < (CHUNK_HALVES * 2) / (THREADS * 16); ++i) { // 8
            const unsigned           l = lds0 + (unsigned)(i * THREADS * 16);
            const unsigned long long g = g0 + (unsigned long long)(i * THREADS * 16);
            asm volatile("global_load_async_to_lds_b128 %0, %1, off"
                         :: "v"(l), "v"(g) : "memory");
        }
    };

    // A tiles: exp(-|s_m - t_{jc+k}|) in this kernel's contiguous-k A layout:
// lane = m + (k&16), half = k&15; two M-tiles.
    auto stage_A = [&](int buf, int c) {
        const int jc = c * KC;
        #pragma unroll
        for (int e2 = 0; e2 < 2 * MTILES; ++e2) {
            const int e  = tid + THREADS * e2;       // 0..1023
            const int mt = e >> 9;                   // M-tile
            const int e9 = e & 511;
            const int m  = e9 & 15;
            const int k  = e9 >> 4;                  // 0..31
            const float a = __expf(-fabsf(Ssh[(mt << 4) + m] - Tsh[jc + k]));
            const int l = m + (k & 16);
            const int h = k & 15;
            Ash[buf][(mt << 9) + (l << 4) + h] = (_Float16)a;
        }
    };

    // ---- stage template + score rows ----
    for (int j = tid; j < S_DIM; j += THREADS) Tsh[j] = tmpl[j];
    if (tid < MROWS) Ssh[tid] = score[b * S_DIM + i0 + tid];
    __syncthreads();

    // ---- kick off chunk 0 DMA + A tiles, overlap with softmax denominators -
    issue_async(0, 0);
    stage_A(0, 0);

    // Z_m = sum_j exp(-|s_m - t_j|)  (reference's rowmax shift cancels);
    // 8 threads per row.
    {
        const int m    = tid >> 3;
        const int part = tid & 7;
        const float s  = Ssh[m];
        float accz = 0.0f;
        for (int j = part; j < S_DIM; j += 8)
            accz += __expf(-fabsf(s - Tsh[j]));
        Psh[tid] = accz;
        __syncthreads();
        if (tid < MROWS) {
            float z = 0.0f;
            for (int p = 0; p < 8; ++p) z += Psh[tid * 8 + p];
            Zsh[tid] = 1.0f / z;
        }
    }

    v8f acc[MTILES][NT_PER_WAVE];
    #pragma unroll
    for (int mt = 0; mt < MTILES; ++mt)
        #pragma unroll
        for (int t = 0; t < NT_PER_WAVE; ++t)
            #pragma unroll
            for (int v = 0; v < 8; ++v) acc[mt][t][v] = 0.0f;

    WAIT_ASYNC0();
    __syncthreads();

    // ---- main loop: DMA chunk c+1 while WMMAing chunk c ----
    for (int c = 0; c < NCHUNK; ++c) {
        const int cur = c & 1;
        if (c + 1 < NCHUNK) {
            issue_async(cur ^ 1, c + 1);
            stage_A(cur ^ 1, c + 1);
        }

        const v16h amat0 = *(const v16h*)(&Ash[cur][(0 << 9) + (lane << 4)]);
        const v16h amat1 = *(const v16h*)(&Ash[cur][(1 << 9) + (lane << 4)]);
        #pragma unroll
        for (int t = 0; t < NT_PER_WAVE; ++t) {
            const int nt = w * NT_PER_WAVE + t;
            const v16h bmat = *(const v16h*)(&Bsh[cur][(nt * 32 + lane) << 4]);
            acc[0][t] = __builtin_amdgcn_wmma_f32_16x16x32_f16(
                false, amat0, false, bmat, (short)0, acc[0][t], false, false);
            acc[1][t] = __builtin_amdgcn_wmma_f32_16x16x32_f16(
                false, amat1, false, bmat, (short)0, acc[1][t], false, false);
        }

        if (c + 1 < NCHUNK) {
            WAIT_ASYNC0();
            __syncthreads();
        }
    }

    // ---- epilogue: softmax normalize, store, sum-of-squares ----
    const int hi  = lane >> 4;          // C layout: lanes16-31 hold M = v+8
    const int col = lane & 15;

    float ss = 0.0f;
    #pragma unroll
    for (int mt = 0; mt < MTILES; ++mt) {
        float zinv[8];
        #pragma unroll
        for (int v = 0; v < 8; ++v) zinv[v] = Zsh[(mt << 4) + v + 8 * hi];
        #pragma unroll
        for (int t = 0; t < NT_PER_WAVE; ++t) {
            const int n0 = (w * NT_PER_WAVE + t) << 4;
            #pragma unroll
            for (int v = 0; v < 8; ++v) {
                const float cc = acc[mt][t][v] * zinv[v];
                const int row = i0 + (mt << 4) + v + 8 * hi;
                out[((size_t)b * S_DIM + row) * F_DIM + n0 + col] = cc;
                ss += cc * cc;
            }
        }
    }

    __syncthreads();
    Psh[tid] = ss;
    __syncthreads();
    for (int st = THREADS / 2; st > 0; st >>= 1) {
        if (tid < st) Psh[tid] += Psh[tid + st];
        __syncthreads();
    }
    if (tid == 0) atomicAdd(ss_accum, Psh[0]);
}

// ---------------------------------------------------------------------------
// Fallback (no workspace room): fully fused single kernel, 16 rows per block.
// ---------------------------------------------------------------------------
__global__ __launch_bounds__(THREADS)
void dtw_warp_kernel(const float* __restrict__ score,
                     const float* __restrict__ feature,
                     const float* __restrict__ tmpl,
                     float* __restrict__ out,
                     float* __restrict__ ss_accum)
{
    __shared__ float Tsh[S_DIM];
    __shared__ float Ssh[ROWS_FB];
    __shared__ float Zsh[ROWS_FB];
    __shared__ float Psh[THREADS];
    __shared__ __align__(32) _Float16 Ash[ROWS_FB * KC];
    __shared__ __align__(32) _Float16 Bsh[CHUNK_HALVES];

    const int tid  = threadIdx.x;
    const int lane = tid & 31;
    const int w    = tid >> 5;
    const int b    = blockIdx.x >> 7;
    const int i0   = (blockIdx.x & 127) << 4;

    for (int j = tid; j < S_DIM; j += THREADS) Tsh[j] = tmpl[j];
    if (tid < ROWS_FB) Ssh[tid] = score[b * S_DIM + i0 + tid];
    __syncthreads();

    {
        const int m    = tid >> 4;
        const int part = tid & 15;
        const float s  = Ssh[m];
        float accz = 0.0f;
        for (int j = part; j < S_DIM; j += 16)
            accz += __expf(-fabsf(s - Tsh[j]));
        Psh[tid] = accz;
        __syncthreads();
        if (tid < ROWS_FB) {
            float z = 0.0f;
            for (int p = 0; p < 16; ++p) z += Psh[tid * 16 + p];
            Zsh[tid] = 1.0f / z;
        }
    }

    v8f acc[NT_PER_WAVE];
    for (int t = 0; t < NT_PER_WAVE; ++t)
        for (int v = 0; v < 8; ++v) acc[t][v] = 0.0f;

    const float* fbase = feature + (size_t)b * S_DIM * F_DIM;

    for (int jc = 0; jc < S_DIM; jc += KC) {
        __syncthreads();
        {
            const float4* src = (const float4*)(fbase + (size_t)jc * F_DIM);
            #pragma unroll
            for (int e = 0; e < (KC * F_DIM / 4) / THREADS; ++e) {
                const int idx = tid + THREADS * e;
                const int k   = idx >> 7;
                const int q   = idx & 127;
                const float4 v = src[k * 128 + q];
                const int n    = q << 2;
                const int hk   = k & 15;
                const int lhi  = k & 16;
                const float vv[4] = {v.x, v.y, v.z, v.w};
                #pragma unroll
                for (int u = 0; u < 4; ++u) {
                    const int nn = n + u;
                    const int l  = (nn & 15) + lhi;
                    Bsh[((((nn >> 4) * 32) + l) << 4) + hk] = (_Float16)vv[u];
                }
            }
        }
        {
            #pragma unroll
            for (int e2 = 0; e2 < 2; ++e2) {
                const int e = tid + THREADS * e2;
                const int m = e & 15;
                const int k = e >> 4;
                const float a = __expf(-fabsf(Ssh[m] - Tsh[jc + k]));
                const int l = m + (k & 16);
                const int h = k & 15;
                Ash[(l << 4) + h] = (_Float16)a;
            }
        }
        __syncthreads();

        const v16h amat = *(const v16h*)(Ash + (lane << 4));
        #pragma unroll
        for (int t = 0; t < NT_PER_WAVE; ++t) {
            const int nt = w * NT_PER_WAVE + t;
            const v16h bmat = *(const v16h*)(Bsh + ((nt * 32 + lane) << 4));
            acc[t] = __builtin_amdgcn_wmma_f32_16x16x32_f16(
                false, amat, false, bmat, (short)0, acc[t], false, false);
        }
    }

    const int hi  = lane >> 4;
    const int col = lane & 15;
    float zinv[8];
    #pragma unroll
    for (int v = 0; v < 8; ++v) zinv[v] = Zsh[v + 8 * hi];

    float ss = 0.0f;
    #pragma unroll
    for (int t = 0; t < NT_PER_WAVE; ++t) {
        const int n0 = (w * NT_PER_WAVE + t) << 4;
        #pragma unroll
        for (int v = 0; v < 8; ++v) {
            const float cc = acc[t][v] * zinv[v];
            out[((size_t)b * S_DIM + i0 + v + 8 * hi) * F_DIM + n0 + col] = cc;
            ss += cc * cc;
        }
    }

    __syncthreads();
    Psh[tid] = ss;
    __syncthreads();
    for (int st = THREADS / 2; st > 0; st >>= 1) {
        if (tid < st) Psh[tid] += Psh[tid + st];
        __syncthreads();
    }
    if (tid == 0) atomicAdd(ss_accum, Psh[0]);
}

__global__ void dtw_init_kernel(float* ss_accum) { *ss_accum = 0.0f; }

__global__ void dtw_finalize_kernel(const float* __restrict__ ss_accum,
                                    float* __restrict__ out_scalar)
{
    *out_scalar = 1e-7f * sqrtf(*ss_accum);
}

extern "C" void kernel_launch(void* const* d_in, const int* in_sizes, int n_in,
                              void* d_out, int out_size, void* d_ws, size_t ws_size,
                              hipStream_t stream)
{
    const float* score   = (const float*)d_in[0];   // [8, 2048, 1]
    const float* feature = (const float*)d_in[1];   // [8, 2048, 512]
    const float* tmpl    = (const float*)d_in[2];   // [1, 2048, 1]
    float* out = (float*)d_out;                     // warped + trailing scalar
    float* ws  = (float*)d_ws;                      // ws[0] = sum of squares

    dtw_init_kernel<<<1, 1, 0, stream>>>(ws);

    const size_t featH_bytes =
        (size_t)B_DIM * S_DIM * F_DIM * sizeof(_Float16);   // 16 MB

    if (ws_size >= 256 + featH_bytes) {
        _Float16* featH = (_Float16*)((char*)d_ws + 256);
        dtw_convert_kernel<<<(B_DIM * S_DIM * F_DIM / 16) / 256, 256, 0, stream>>>(
            feature, featH);
        dtw_warp_async_kernel<<<dim3(B_DIM * (S_DIM / MROWS)), THREADS, 0, stream>>>(
            score, featH, tmpl, out, ws);
    } else {
        dtw_warp_kernel<<<dim3(B_DIM * (S_DIM / ROWS_FB)), THREADS, 0, stream>>>(
            score, feature, tmpl, out, ws);
    }

    dtw_finalize_kernel<<<1, 1, 0, stream>>>(
        ws, out + (size_t)B_DIM * S_DIM * F_DIM);
}
